// OutputHeads_8899172238093
// MI455X (gfx1250) — matrix operand order repaired and checked by reading, not compiled, not run on hardware
//
#include <hip/hip_runtime.h>
#include <hip/hip_bf16.h>

// ---------------------------------------------------------------------------
// Types for CDNA5 WMMA
// ---------------------------------------------------------------------------
typedef __attribute__((ext_vector_type(16))) __bf16 v16bf;
typedef __attribute__((ext_vector_type(8)))  float  v8f;

#define N_PROPOSALS 4096
#define NUM_CLASSES 91
#define FEAT        256
#define D_IN        (256 * 7 * 7)   // 12544
#define HEAD_N      512             // padded: cls at [0,91), reg at [128,492)
#define REG_OFF     128
#define IMG_W       1280.0f
#define IMG_H       384.0f
#define SCORE_THR   0.05f
#define IOU_THR     0.5f

#define KC          128             // K-chunk staged to LDS per stage
#define LDS_PITCH   132             // 128 + 4 floats pad: kills LDS bank conflicts

// ---------------------------------------------------------------------------
// CDNA5 async global->LDS copy (ASYNCcnt-tracked) + wait, via inline asm
// (CDNA5 ISA). VDST = LDS byte offset VGPR, VADDR = 64-bit
// global address VGPR pair (GV mode). The low 32 bits of a flat shared
// pointer are the LDS byte offset (ISA §10.2 aperture rules).
// ---------------------------------------------------------------------------
__device__ __forceinline__ void async_copy_b128(const void* g, void* l) {
    unsigned           lo = (unsigned)(unsigned long long)l;   // LDS byte offset
    unsigned long long ga = (unsigned long long)g;             // global address
    asm volatile("global_load_async_to_lds_b128 %0, %1, off"
                 :: "v"(lo), "v"(ga) : "memory");
}

__device__ __forceinline__ void wait_async0() {
#if __has_builtin(__builtin_amdgcn_s_wait_asynccnt)
    __builtin_amdgcn_s_wait_asynccnt(0);
#else
    asm volatile("s_wait_asynccnt 0x0" ::: "memory");
#endif
}

// ---------------------------------------------------------------------------
// Prep: W1 [12544,256] f32 row-major  ->  W1t [256][12544] bf16 column-major
// ---------------------------------------------------------------------------
__global__ void prep_w1t_kernel(const float* __restrict__ W1,
                                __bf16* __restrict__ W1t) {
    int idx = blockIdx.x * blockDim.x + threadIdx.x;   // FEAT * D_IN
    if (idx >= FEAT * D_IN) return;
    int col = idx / D_IN;
    int k   = idx % D_IN;
    W1t[idx] = (__bf16)W1[(size_t)k * FEAT + col];
}

// ---------------------------------------------------------------------------
// Prep: fused, padded head weights, column-major bf16 [HEAD_N][FEAT],
// plus padded bias (f32[HEAD_N]).
// ---------------------------------------------------------------------------
__global__ void prep_heads_kernel(const float* __restrict__ Wc,
                                  const float* __restrict__ bc,
                                  const float* __restrict__ Wr,
                                  const float* __restrict__ br,
                                  __bf16* __restrict__ Wcatt,
                                  float* __restrict__ biascat) {
    int idx = blockIdx.x * blockDim.x + threadIdx.x;   // HEAD_N * FEAT
    if (idx >= HEAD_N * FEAT) return;
    int col = idx / FEAT;
    int k   = idx % FEAT;
    float v = 0.0f;
    if (col < NUM_CLASSES)
        v = Wc[(size_t)k * NUM_CLASSES + col];
    else if (col >= REG_OFF && col < REG_OFF + 4 * NUM_CLASSES)
        v = Wr[(size_t)k * (4 * NUM_CLASSES) + (col - REG_OFF)];
    Wcatt[idx] = (__bf16)v;
    if (k == 0) {
        float b = 0.0f;
        if (col < NUM_CLASSES) b = bc[col];
        else if (col >= REG_OFF && col < REG_OFF + 4 * NUM_CLASSES) b = br[col - REG_OFF];
        biascat[col] = b;
    }
}

// ---------------------------------------------------------------------------
// GEMM: C[M,N] = A[M,K](f32) * B[K,N] + bias, via bf16 WMMA, f32 accumulate.
//  - A tile (16 rows x 128 K, f32) is DMA'd to LDS with async b128 copies,
//    double-buffered: stage s+1 streams in while stage s feeds the WMMAs.
//    One copy per block (shared by all 8 waves) instead of 8 redundant loads.
//  - B is pre-transposed column-major bf16, so each lane's 32x16 B fragment
//    (16 consecutive K at fixed col) is one aligned 32B load.
//  - N == NT*8*16 exactly; no guards -> EXEC-full WMMA loop, no reg churn.
// C/D layout per CDNA5 ISA §7.12.2; A/B use contiguous k = 16*h + s.
// ---------------------------------------------------------------------------
template <int NT>
__global__ void gemm_wmma_kernel(const float* __restrict__ A,
                                 const __bf16* __restrict__ Bt,
                                 const float* __restrict__ bias,
                                 float* __restrict__ C,
                                 int N, int K) {
    __shared__ float smem[2][16][LDS_PITCH];

    const int tid  = threadIdx.x;
    const int wave = tid >> 5;
    const int lane = tid & 31;
    const int hb   = lane >> 4;
    const int l16  = lane & 15;
    const int tileM = blockIdx.x;
    const float* Ablock = A + (size_t)tileM * 16 * K;

    int cb[NT];
    const __bf16* Bp[NT];
    v8f acc[NT] = {};
    #pragma unroll
    for (int t = 0; t < NT; ++t) {
        cb[t] = (wave + t * 8) * 16 + l16;
        Bp[t] = Bt + (size_t)cb[t] * K + hb * 16;
    }

    // issue one stage: 16 rows x KC floats = 512 x 16B chunks, 2 per thread
    auto issue_stage = [&](int buf, int k0) {
        #pragma unroll
        for (int u = 0; u < 2; ++u) {
            int c   = tid + u * 256;
            int row = c >> 5;          // 32 chunks per row
            int kk4 = c & 31;          // chunk within row (4 floats each)
            const float* g = Ablock + (size_t)row * K + k0 + kk4 * 4;
            async_copy_b128(g, &smem[buf][row][kk4 * 4]);
        }
    };

    const int nstages = K / KC;
    issue_stage(0, 0);

    for (int s = 0; s < nstages; ++s) {
        const int buf = s & 1;
        wait_async0();        // this wave's stage-s transfers complete
        __syncthreads();      // => every wave's stage-s transfers visible
        if (s + 1 < nstages)  // overlap next DMA with this stage's WMMAs
            issue_stage(buf ^ 1, (s + 1) * KC);

        const int k0 = s * KC;
        #pragma unroll
        for (int ks = 0; ks < KC; ks += 32) {
            // A fragment from LDS (one contiguous 16-float run per lane)
            const float* As = &smem[buf][l16][ks + hb * 16];
            v16bf a;
            #pragma unroll
            for (int e = 0; e < 8; ++e)  a[e] = (__bf16)As[e];
            #pragma unroll
            for (int e = 8; e < 16; ++e) a[e] = (__bf16)As[e];

            const int kg = k0 + ks;
            #pragma unroll
            for (int t = 0; t < NT; ++t) {
                v16bf b = *reinterpret_cast<const v16bf*>(Bp[t] + kg);
                acc[t] = __builtin_amdgcn_wmma_f32_16x16x32_bf16(
                    false, a, false, b, (short)0, acc[t], false, false);
            }
        }
        __builtin_prefetch(Bp[0] + k0 + KC, 0, 1);   // keep B stream warm
    }

    #pragma unroll
    for (int t = 0; t < NT; ++t) {
        const float bv = bias[cb[t]];
        #pragma unroll
        for (int r = 0; r < 8; ++r) {
            int row = tileM * 16 + hb * 8 + r;
            C[(size_t)row * N + cb[t]] = acc[t][r] + bv;
        }
    }
}

// ---------------------------------------------------------------------------
// Per-proposal: argmax label, softmax probits, score, pass flag, box decode.
// Reads logits/deltas from the fused padded head buffer (stride HEAD_N).
// Replicates the reference verbatim, incl. cx = py1 + pw/2, cy = px1 + ph/2.
// ---------------------------------------------------------------------------
__global__ void decode_softmax_kernel(const float* __restrict__ head,
                                      const float* __restrict__ proposals,
                                      float* __restrict__ probits,
                                      float* __restrict__ boxes,
                                      float* __restrict__ scores,
                                      int* __restrict__ passes) {
    int n = blockIdx.x * blockDim.x + threadIdx.x;
    if (n >= N_PROPOSALS) return;

    const float* lr = head + (size_t)n * HEAD_N;   // logits at offset 0
    float best = lr[0];
    int label = 0;
    for (int c = 1; c < NUM_CLASSES; ++c) {
        float v = lr[c];
        if (v > best) { best = v; label = c; }   // first-max, like jnp.argmax
    }
    float sum = 0.0f;
    for (int c = 0; c < NUM_CLASSES; ++c) sum += expf(lr[c] - best);
    float inv = 1.0f / sum;
    bool pass = false;
    for (int c = 0; c < NUM_CLASSES; ++c) {
        float p = expf(lr[c] - best) * inv;
        probits[(size_t)n * NUM_CLASSES + c] = p;
        if (p > SCORE_THR) pass = true;
    }
    scores[n] = inv;      // max probit == exp(0)/sum
    passes[n] = pass ? 1 : 0;

    const float* d = lr + REG_OFF + label * 4;     // deltas at offset 128
    float dx = d[0], dy = d[1], dw = d[2], dh = d[3];
    float px1 = proposals[n * 4 + 0], py1 = proposals[n * 4 + 1];
    float px2 = proposals[n * 4 + 2], py2 = proposals[n * 4 + 3];
    float pw = px2 - px1;
    float ph = py2 - py1;
    float cx = py1 + pw * 0.5f;   // reference quirk: uses py1 with pw
    float cy = px1 + ph * 0.5f;   // reference quirk: uses px1 with ph
    float gx = cx + dx * pw;
    float gy = cy + dy * ph;
    float gw = pw * expf(dw);
    float gh = ph * expf(dh);
    float bx1 = gx - gw * 0.5f, by1 = gy - gh * 0.5f;
    float bx2 = gx + gw * 0.5f, by2 = gy + gh * 0.5f;
    bx1 = fminf(fmaxf(bx1, 0.0f), IMG_W);
    by1 = fminf(fmaxf(by1, 0.0f), IMG_H);
    bx2 = fminf(fmaxf(bx2, 0.0f), IMG_W);
    by2 = fminf(fmaxf(by2, 0.0f), IMG_H);
    boxes[n * 4 + 0] = bx1;
    boxes[n * 4 + 1] = by1;
    boxes[n * 4 + 2] = bx2;
    boxes[n * 4 + 3] = by2;
}

// ---------------------------------------------------------------------------
// Bitonic sort of 4096 (score, idx) pairs: descending score, ascending idx on
// ties — matches jnp.argsort(-scores) (stable). Single block, 1024 threads.
// ---------------------------------------------------------------------------
__global__ void sort_kernel(const float* __restrict__ scores,
                            int* __restrict__ order) {
    __shared__ float ss[N_PROPOSALS];
    __shared__ int   si[N_PROPOSALS];
    const int tid = threadIdx.x;
    for (int i = tid; i < N_PROPOSALS; i += 1024) { ss[i] = scores[i]; si[i] = i; }
    __syncthreads();
    for (int size = 2; size <= N_PROPOSALS; size <<= 1) {
        for (int stride = size >> 1; stride > 0; stride >>= 1) {
            for (int t = tid; t < N_PROPOSALS / 2; t += 1024) {
                int lo = 2 * t - (t & (stride - 1));
                int hi = lo + stride;
                bool dirF = ((lo & size) == 0);
                float sl = ss[lo], sh = ss[hi];
                int   il = si[lo], ih = si[hi];
                bool hiBefore = (sh > sl) || (sh == sl && ih < il);
                if (hiBefore == dirF) {
                    ss[lo] = sh; ss[hi] = sl;
                    si[lo] = ih; si[hi] = il;
                }
            }
            __syncthreads();
        }
    }
    for (int i = tid; i < N_PROPOSALS; i += 1024) order[i] = si[i];
}

// ---------------------------------------------------------------------------
// Gather boxes/passes into sorted order.
// ---------------------------------------------------------------------------
__global__ void gather_kernel(const float* __restrict__ boxes,
                              const int* __restrict__ passes,
                              const int* __restrict__ order,
                              float* __restrict__ boxes_s,
                              int* __restrict__ passes_s) {
    int i = blockIdx.x * blockDim.x + threadIdx.x;
    if (i >= N_PROPOSALS) return;
    int src = order[i];
    boxes_s[i * 4 + 0] = boxes[src * 4 + 0];
    boxes_s[i * 4 + 1] = boxes[src * 4 + 1];
    boxes_s[i * 4 + 2] = boxes[src * 4 + 2];
    boxes_s[i * 4 + 3] = boxes[src * 4 + 3];
    passes_s[i] = passes[src];
}

// ---------------------------------------------------------------------------
// Parallel NMS, stage 1: suppression bitmask.
// mask[i][w] bit b set  <=>  j = w*32+b, j < i, IoU(box_i, box_j) > 0.5.
// Fully parallel: 4096 blocks x 128 threads, 32 IoUs per thread.
// ---------------------------------------------------------------------------
__global__ void nms_mask_kernel(const float* __restrict__ bs,
                                unsigned* __restrict__ mask) {
    const int i = blockIdx.x;
    const int w = threadIdx.x;            // 0..127

    float x1i = bs[i * 4 + 0], y1i = bs[i * 4 + 1];
    float x2i = bs[i * 4 + 2], y2i = bs[i * 4 + 3];
    float ai = (x2i - x1i) * (y2i - y1i);

    unsigned m = 0u;
    int jbase = w * 32;
    #pragma unroll 4
    for (int b = 0; b < 32; ++b) {
        int j = jbase + b;
        if (j < i) {
            float x1j = bs[j * 4 + 0], y1j = bs[j * 4 + 1];
            float x2j = bs[j * 4 + 2], y2j = bs[j * 4 + 3];
            float aj = (x2j - x1j) * (y2j - y1j);
            float lx = fmaxf(x1i, x1j), ly = fmaxf(y1i, y1j);
            float rx = fminf(x2i, x2j), ry = fminf(y2i, y2j);
            float ww = fmaxf(rx - lx, 0.0f), hh = fmaxf(ry - ly, 0.0f);
            float inter = ww * hh;
            float iou = inter / (ai + aj - inter);
            if (iou > IOU_THR) m |= (1u << b);   // NaN compares false (matches JAX)
        }
    }
    mask[(size_t)i * (N_PROPOSALS / 32) + w] = m;
}

// ---------------------------------------------------------------------------
// Parallel NMS, stage 2: sequential greedy scan over the bitmask with a
// single wave32 — no barriers, keep-set held in 4 regs/lane, wave vote __any.
// ---------------------------------------------------------------------------
__global__ void nms_scan_kernel(const unsigned* __restrict__ mask,
                                const int* __restrict__ ps,
                                int* __restrict__ keep) {
    const int lane = threadIdx.x;         // 0..31
    unsigned kb[4] = {0u, 0u, 0u, 0u};    // lane owns words lane + 32*q

    for (int i = 0; i < N_PROPOSALS; ++i) {
        const unsigned* mrow = mask + (size_t)i * (N_PROPOSALS / 32);
        bool found = false;
        #pragma unroll
        for (int q = 0; q < 4; ++q) {
            unsigned mw = mrow[lane + 32 * q];
            if (mw & kb[q]) found = true;
        }
        bool sup = __any(found);
        int kp = (ps[i] != 0 && !sup) ? 1 : 0;
        int w = i >> 5;
        if (kp && lane == (w & 31)) kb[w >> 5] |= (1u << (i & 31));
        if (lane == 0) keep[i] = kp;
    }
}

// ---------------------------------------------------------------------------
// Exclusive prefix sum of keep[] (stable-compaction destinations).
// ---------------------------------------------------------------------------
__global__ void scan_kernel(const int* __restrict__ keep,
                            int* __restrict__ pref) {
    __shared__ int partial[1024];
    const int tid = threadIdx.x;
    int v[4];
    int s = 0;
    for (int u = 0; u < 4; ++u) { v[u] = keep[tid * 4 + u]; s += v[u]; }
    partial[tid] = s;
    __syncthreads();
    for (int off = 1; off < 1024; off <<= 1) {
        int t = (tid >= off) ? partial[tid - off] : 0;
        __syncthreads();
        partial[tid] += t;
        __syncthreads();
    }
    int run = (tid > 0) ? partial[tid - 1] : 0;
    for (int u = 0; u < 4; ++u) { pref[tid * 4 + u] = run; run += v[u]; }
}

// ---------------------------------------------------------------------------
// Zero-init full output, then scatter kept rows.
// Output layout: probits[4096*91] | boxes[4096*4] | valid[4096]
// ---------------------------------------------------------------------------
__global__ void zero_kernel(float* __restrict__ out, int n) {
    int i = blockIdx.x * blockDim.x + threadIdx.x;
    if (i < n) out[i] = 0.0f;
}

__global__ void scatter_kernel(const float* __restrict__ probits,
                               const float* __restrict__ boxes_s,
                               const int* __restrict__ order,
                               const int* __restrict__ keep,
                               const int* __restrict__ pref,
                               float* __restrict__ out) {
    const int i = blockIdx.x;
    if (!keep[i]) return;
    const int dest = pref[i];
    const int src = order[i];
    float* out_prob  = out;
    float* out_box   = out + (size_t)N_PROPOSALS * NUM_CLASSES;
    float* out_valid = out_box + (size_t)N_PROPOSALS * 4;
    for (int c = threadIdx.x; c < NUM_CLASSES; c += blockDim.x)
        out_prob[(size_t)dest * NUM_CLASSES + c] = probits[(size_t)src * NUM_CLASSES + c];
    if (threadIdx.x < 4)
        out_box[dest * 4 + threadIdx.x] = boxes_s[i * 4 + threadIdx.x];
    if (threadIdx.x == 0)
        out_valid[dest] = 1.0f;
}

// ---------------------------------------------------------------------------
// Host-side launch
// ---------------------------------------------------------------------------
extern "C" void kernel_launch(void* const* d_in, const int* in_sizes, int n_in,
                              void* d_out, int out_size, void* d_ws, size_t ws_size,
                              hipStream_t stream) {
    const float* pooled    = (const float*)d_in[0];   // 4096*256*7*7
    const float* proposals = (const float*)d_in[1];   // 4096*4
    const float* W1        = (const float*)d_in[2];   // 12544*256
    const float* b1        = (const float*)d_in[3];   // 256
    const float* Wc        = (const float*)d_in[4];   // 256*91
    const float* bc        = (const float*)d_in[5];   // 91
    const float* Wr        = (const float*)d_in[6];   // 256*364
    const float* br        = (const float*)d_in[7];   // 364
    float* out = (float*)d_out;

    // ---- workspace carve-up (f32 region, then bf16, then int) ----
    float* feat    = (float*)d_ws;                                // 4096*256
    float* head    = feat    + (size_t)N_PROPOSALS * FEAT;        // 4096*512
    float* probits = head    + (size_t)N_PROPOSALS * HEAD_N;      // 4096*91
    float* boxes   = probits + (size_t)N_PROPOSALS * NUM_CLASSES; // 4096*4
    float* scores  = boxes   + (size_t)N_PROPOSALS * 4;           // 4096
    float* boxes_s = scores  + N_PROPOSALS;                       // 4096*4
    float* biascat = boxes_s + (size_t)N_PROPOSALS * 4;           // 512
    __bf16* W1t    = (__bf16*)(biascat + HEAD_N);                 // 256*12544
    __bf16* Wcatt  = W1t + (size_t)FEAT * D_IN;                   // 512*256
    unsigned* mask = (unsigned*)(Wcatt + (size_t)HEAD_N * FEAT);  // 4096*128
    int*   passes  = (int*)(mask + (size_t)N_PROPOSALS * (N_PROPOSALS / 32));
    int*   order   = passes + N_PROPOSALS;
    int*   ps_s    = order  + N_PROPOSALS;
    int*   keep    = ps_s   + N_PROPOSALS;
    int*   pref    = keep   + N_PROPOSALS;

    // 0) weight prep: transpose (+pad, +fuse) to column-major bf16
    prep_w1t_kernel<<<(FEAT * D_IN + 255) / 256, 256, 0, stream>>>(W1, W1t);
    prep_heads_kernel<<<(HEAD_N * FEAT + 255) / 256, 256, 0, stream>>>(
        Wc, bc, Wr, br, Wcatt, biascat);

    // 1) feat = pooled @ W1 + b1   (M=4096, K=12544, N=256; 16 tiles, NT=2)
    gemm_wmma_kernel<2><<<N_PROPOSALS / 16, 256, 0, stream>>>(
        pooled, W1t, b1, feat, FEAT, D_IN);

    // 2) head = feat @ [Wc|pad|Wr|pad] + bias  (N=512; 32 tiles, NT=4)
    gemm_wmma_kernel<4><<<N_PROPOSALS / 16, 256, 0, stream>>>(
        feat, Wcatt, biascat, head, HEAD_N, FEAT);

    // 3) decode + softmax + pass flags
    decode_softmax_kernel<<<N_PROPOSALS / 256, 256, 0, stream>>>(
        head, proposals, probits, boxes, scores, passes);

    // 4) sort by descending score (stable wrt index)
    sort_kernel<<<1, 1024, 0, stream>>>(scores, order);

    // 5) gather boxes & passes into sorted order
    gather_kernel<<<N_PROPOSALS / 256, 256, 0, stream>>>(
        boxes, passes, order, boxes_s, ps_s);

    // 6) parallel suppression bitmask, then single-wave greedy scan
    nms_mask_kernel<<<N_PROPOSALS, N_PROPOSALS / 32, 0, stream>>>(boxes_s, mask);
    nms_scan_kernel<<<1, 32, 0, stream>>>(mask, ps_s, keep);

    // 7) stable compaction destinations
    scan_kernel<<<1, 1024, 0, stream>>>(keep, pref);

    // 8) zero output, scatter kept rows
    zero_kernel<<<(out_size + 255) / 256, 256, 0, stream>>>(out, out_size);
    scatter_kernel<<<N_PROPOSALS, 96, 0, stream>>>(
        probits, boxes_s, order, keep, pref, out);
}
